// PVRNN_41893111005641
// MI455X (gfx1250) — compile-verified, matrix operand order checked by reading, not run on hardware
//
#include <hip/hip_runtime.h>

// ------------------------------------------------------------------
// PVRNN forward for MI455X (gfx1250, wave32).
// All matmuls via v_wmma_f32_16x16x32_bf16 (bf16 operands, f32 acc).
// Branch-free, software-pipelined inner loop: next K-step's A/B
// fragments are loaded while the current step's 4 WMMAs execute,
// so B-load latency is hidden within a wave (critical at the low
// occupancy of the M=128 recurrent-step GEMMs).
// Weights (~11MB bf16) stay resident in the 192MB L2 across the
// 64 strictly-sequential timesteps.
// ------------------------------------------------------------------

#define B_    128
#define T_    64
#define L_    2
#define H_    512
#define S_    256
#define CL_   16
#define CS_   32
#define NOBJ_ 5
#define OS_   32
#define A_    16

typedef __attribute__((ext_vector_type(16))) __bf16 v16bf;
typedef __attribute__((ext_vector_type(8)))  __bf16 v8bf;
typedef __attribute__((ext_vector_type(8)))  float  v8f;

// One K-segment of a (virtually concatenated) GEMM input. K % 32 == 0 always.
struct Seg {
  const float*  a;   // fp32 activation base
  long sB, sT;       // row m address = a + (m/TT)*sB + (m%TT)*sT
  int  TT;
  int  K;            // segment K extent (multiple of 32)
  const __bf16* wt;  // transposed weights: (padded) N rows of length wK, bf16
  int  wK;           // row length (multiple of 32)
  int  k0;           // column offset into wt rows
};

struct GemmArgs {
  Seg seg[5]; int nseg; int N;
  // column split (merged mu/std heads): n < split -> out/bias/act, else out2/bias2/act2
  int split;
  // row pairing (merged hp/hq): group = m >> pairShift, out += group*outB
  int pairShift; long outB;
  const float* bias;  const float* bias2;
  const float* addmat; long addS;          // optional pre-activation add (by mm,nc)
  const float* blend;  long blendS;        // optional MTRNN blend source (by mm,nc)
  float alpha, beta;                       // out = alpha*blend + beta*act(acc)
  float* out; float* out2; long outS;
  int act, act2;                           // 0 none 1 prelu 2 tanh 3 softplus
};

__device__ __forceinline__ v16bf packA(float4 a0, float4 a1, float4 a2, float4 a3) {
  v16bf f;
  f[0]=(__bf16)a0.x; f[1]=(__bf16)a0.y; f[2]=(__bf16)a0.z; f[3]=(__bf16)a0.w;
  f[4]=(__bf16)a1.x; f[5]=(__bf16)a1.y; f[6]=(__bf16)a1.z; f[7]=(__bf16)a1.w;
  f[8]=(__bf16)a2.x; f[9]=(__bf16)a2.y; f[10]=(__bf16)a2.z; f[11]=(__bf16)a2.w;
  f[12]=(__bf16)a3.x; f[13]=(__bf16)a3.y; f[14]=(__bf16)a3.z; f[15]=(__bf16)a3.w;
  return f;
}
__device__ __forceinline__ v16bf packB(v8bf b0, v8bf b1) {
  v16bf f;
#pragma unroll
  for (int j = 0; j < 8; ++j) { f[j] = b0[j]; f[j + 8] = b1[j]; }
  return f;
}

// Block = 128 threads = 4 waves. Wave tile: 16(M) x 64(N), 4 accumulators.
__global__ __launch_bounds__(128) void gemm_wmma(GemmArgs g) {
  const int lane = threadIdx.x & 31;
  const int wave = threadIdx.x >> 5;
  const int m0   = blockIdx.y << 4;
  const int n0   = (blockIdx.x << 8) + (wave << 6);
  const int mr   = lane & 15;
  const int hi   = lane >> 4;
  const int koff = hi << 3;
  const int m    = m0 + mr;

  v8f acc[4];
#pragma unroll
  for (int s = 0; s < 4; ++s) acc[s] = v8f{};

  for (int s = 0; s < g.nseg; ++s) {
    const Seg sg = g.seg[s];
    // CDNA5 16-bit fragment layout: lane row/col = lane&15; lanes<16 carry
    // K {0..7,16..23}, lanes>=16 carry K {8..15,24..31}.
    const float* arow =
        sg.a + (long)(m / sg.TT) * sg.sB + (long)(m % sg.TT) * sg.sT + koff;
    const long wK = sg.wK;
    const __bf16* wb = sg.wt + sg.k0 + koff;
    const __bf16* w0 = wb + (long)(n0 +  0 + mr) * wK;
    const __bf16* w1 = wb + (long)(n0 + 16 + mr) * wK;
    const __bf16* w2 = wb + (long)(n0 + 32 + mr) * wK;
    const __bf16* w3 = wb + (long)(n0 + 48 + mr) * wK;

    // ---- pipeline preload (kk = 0) ----
    float4 a0 = *(const float4*)(arow);
    float4 a1 = *(const float4*)(arow + 4);
    float4 a2 = *(const float4*)(arow + 16);
    float4 a3 = *(const float4*)(arow + 20);
    v8bf b00 = *(const v8bf*)(w0),      b01 = *(const v8bf*)(w0 + 16);
    v8bf b10 = *(const v8bf*)(w1),      b11 = *(const v8bf*)(w1 + 16);
    v8bf b20 = *(const v8bf*)(w2),      b21 = *(const v8bf*)(w2 + 16);
    v8bf b30 = *(const v8bf*)(w3),      b31 = *(const v8bf*)(w3 + 16);

    for (int kk = 0; kk < sg.K; kk += 32) {
      // issue next iteration's loads first (tail re-loads offset 0, discarded)
      const long nk = (kk + 32 < sg.K) ? (long)(kk + 32) : 0;
      float4 na0 = *(const float4*)(arow + nk);
      float4 na1 = *(const float4*)(arow + nk + 4);
      float4 na2 = *(const float4*)(arow + nk + 16);
      float4 na3 = *(const float4*)(arow + nk + 20);
      v8bf nb00 = *(const v8bf*)(w0 + nk), nb01 = *(const v8bf*)(w0 + nk + 16);
      v8bf nb10 = *(const v8bf*)(w1 + nk), nb11 = *(const v8bf*)(w1 + nk + 16);
      v8bf nb20 = *(const v8bf*)(w2 + nk), nb21 = *(const v8bf*)(w2 + nk + 16);
      v8bf nb30 = *(const v8bf*)(w3 + nk), nb31 = *(const v8bf*)(w3 + nk + 16);

      // consume current registers
      v16bf af = packA(a0, a1, a2, a3);
      acc[0] = __builtin_amdgcn_wmma_f32_16x16x32_bf16(
          false, af, false, packB(b00, b01), (short)0, acc[0], false, false);
      acc[1] = __builtin_amdgcn_wmma_f32_16x16x32_bf16(
          false, af, false, packB(b10, b11), (short)0, acc[1], false, false);
      acc[2] = __builtin_amdgcn_wmma_f32_16x16x32_bf16(
          false, af, false, packB(b20, b21), (short)0, acc[2], false, false);
      acc[3] = __builtin_amdgcn_wmma_f32_16x16x32_bf16(
          false, af, false, packB(b30, b31), (short)0, acc[3], false, false);

      // rotate pipeline
      a0 = na0; a1 = na1; a2 = na2; a3 = na3;
      b00 = nb00; b01 = nb01; b10 = nb10; b11 = nb11;
      b20 = nb20; b21 = nb21; b30 = nb30; b31 = nb31;
    }
  }

#pragma unroll
  for (int sub = 0; sub < 4; ++sub) {
    const int n = n0 + (sub << 4) + mr;
    if (n >= g.N) continue;
    const bool second = (g.split != 0) && (n >= g.split);
    const float* bp = second ? g.bias2 : g.bias;
    float*       ob = second ? g.out2  : g.out;
    const int  actc = second ? g.act2  : g.act;
    const int    nc = second ? n - g.split : n;
    const float  bn = bp ? bp[nc] : 0.0f;
#pragma unroll
    for (int r = 0; r < 8; ++r) {
      const int mA = m0 + r + (hi << 3);
      int grp = 0, mm = mA;
      if (g.pairShift) { grp = mA >> g.pairShift; mm = mA & ((1 << g.pairShift) - 1); }
      float v = acc[sub][r] + bn;
      if (g.addmat) v += g.addmat[(long)mm * g.addS + nc];
      if (actc == 1)      v = (v >= 0.0f) ? v : 0.25f * v;
      else if (actc == 2) v = tanhf(v);
      else if (actc == 3) v = (v > 20.0f) ? v : log1pf(expf(v));
      if (g.blend) v = g.alpha * g.blend[(long)mm * g.blendS + nc] + g.beta * v;
      ob[(long)grp * g.outB + (long)mm * g.outS + nc] = v;
    }
  }
}

// fp32 (K,N) -> bf16 transposed (Npad x Kpad), zero padded.
__global__ void wt_convert(const float* __restrict__ w, __bf16* __restrict__ wt,
                           int K, int N, int Kpad, int Npad) {
  int i = blockIdx.x * blockDim.x + threadIdx.x;
  if (i >= Kpad * Npad) return;
  int n = i / Kpad, k = i - n * Kpad;
  wt[i] = (k < K && n < N) ? (__bf16)w[(long)k * N + n] : (__bf16)0.0f;
}

__global__ void add_bias(const float* a, const float* b, float* o, int n) {
  int i = blockIdx.x * blockDim.x + threadIdx.x;
  if (i < n) o[i] = a[i] + b[i];
}

// zero-pad actions (B*T,16) -> (B*T,32)
__global__ void pad_act(const float* __restrict__ a, float* __restrict__ o) {
  int i = blockIdx.x * blockDim.x + threadIdx.x;
  if (i >= B_ * T_ * 32) return;
  int r = i >> 5, c = i & 31;
  o[i] = (c < A_) ? a[(long)r * A_ + c] : 0.0f;
}

// comm mask: nz per row (any of CS nonzero), suffix-OR over CL, multiply.
__global__ void mask_comm(const float* __restrict__ src, float* __restrict__ dst) {
  int wid  = (blockIdx.x * blockDim.x + threadIdx.x) >> 5;
  int lane = threadIdx.x & 31;
  if (wid >= B_ * T_) return;
  const float* s = src + (size_t)wid * CL_ * CS_;
  float*       d = dst + (size_t)wid * CL_ * CS_;
  bool nz = false;
  if (lane < CL_)
    for (int i = 0; i < CS_; ++i) nz |= (s[lane * CS_ + i] != 0.0f);
  unsigned bits = (unsigned)__ballot(nz) & 0xFFFFu;
  if (lane < CL_) {
    float mv = ((bits >> lane) != 0u) ? 1.0f : 0.0f;
    for (int i = 0; i < CS_; ++i) d[lane * CS_ + i] = s[lane * CS_ + i] * mv;
  }
}

// z[0:B*S] = mu_p + eps_p*std_p ; z[B*S:2B*S] = mu_q + eps_q*std_q
__global__ void make_z2(const float* mup, const float* sdp,
                        const float* muq, const float* sdq, long muS,
                        const float* epsP, const float* epsQ, float* z) {
  int i = blockIdx.x * blockDim.x + threadIdx.x;
  if (i >= 2 * B_ * S_) return;
  int half = i >= B_ * S_;
  int j = i - half * B_ * S_;
  int b = j >> 8, s = j & 255;
  const float* mu = half ? muq : mup;
  const float* sd = half ? sdq : sdp;
  const float* ep = half ? epsQ : epsP;
  z[i] = mu[(long)b * muS + s] + ep[j] * sd[(long)b * muS + s];
}

extern "C" void kernel_launch(void* const* d_in, const int* in_sizes, int n_in,
                              void* d_out, int out_size, void* d_ws, size_t ws_size,
                              hipStream_t stream) {
  (void)in_sizes; (void)n_in; (void)out_size; (void)ws_size;
  const float* prevh        = (const float*)d_in[0];
  const float* objects      = (const float*)d_in[1];
  const float* comms_in     = (const float*)d_in[2];
  const float* prev_actions = (const float*)d_in[3];
  const float* prev_cout    = (const float*)d_in[4];
  const float* eps_p        = (const float*)d_in[5];
  const float* eps_q        = (const float*)d_in[6];
  // 0 zp_mu0 1 zp_std0 2 zq_mu0 3 zq_std0 4 Wx0 5 Wh0 6 zp_mu1 7 zp_std1
  // 8 zq_mu1 9 zq_std1 10 Wx1 11 Wh1 12 W_obj 13 W_cobs 14 W_act 15 W_cin
  // 16 W_po 17 W_pc
  const float *Wm[18], *Bv[18];
  for (int i = 0; i < 18; ++i) { Wm[i] = (const float*)d_in[7 + 2 * i]; Bv[i] = (const float*)d_in[8 + 2 * i]; }

  // ---- output slabs (floats) ----
  float* out     = (float*)d_out;
  float* zp_mu   = out;                 // (B,T,L,S)
  float* zp_std  = out + 4194304;
  float* zq_mu   = out + 8388608;
  float* zq_std  = out + 12582912;
  float* hp      = out + 16777216;      // (B,T,L,H)
  float* hq      = out + 25165824;
  float* pred_o  = out + 33554432;      // (B,63,160)
  float* pred_c  = out + 34844672;      // (B,63,512)
  const long hpq = 8388608;             // hq - hp (floats)

  // ---- workspace bump allocator ----
  char* wsb = (char*)d_ws; size_t off = 0;
  auto alloc = [&](size_t bytes) -> void* {
    off = (off + 255) & ~(size_t)255; void* p = wsb + off; off += bytes; return p;
  };
  auto conv = [&](const float* src, __bf16* dst, int K, int N, int Kpad, int Npad) {
    int tot = Kpad * Npad;
    wt_convert<<<(tot + 255) / 256, 256, 0, stream>>>(src, dst, K, N, Kpad, Npad);
  };
  // merged mu/std head weights: mu rows [0,256), std rows [256,512)
  __bf16* WTzp0 = (__bf16*)alloc((size_t)1536 * 512 * 2);
  __bf16* WTzq0 = (__bf16*)alloc((size_t)2560 * 512 * 2);
  __bf16* WTzp1 = (__bf16*)alloc((size_t)512  * 512 * 2);
  __bf16* WTzq1 = (__bf16*)alloc((size_t)1024 * 512 * 2);
  __bf16* WTx0  = (__bf16*)alloc((size_t)768  * 512 * 2);
  __bf16* WTh0  = (__bf16*)alloc((size_t)512  * 512 * 2);
  __bf16* WTx1  = (__bf16*)alloc((size_t)256  * 512 * 2);
  __bf16* WTh1  = (__bf16*)alloc((size_t)512  * 512 * 2);
  __bf16* WTobj = (__bf16*)alloc((size_t)160  * 512 * 2);
  __bf16* WTcob = (__bf16*)alloc((size_t)512  * 512 * 2);
  __bf16* WTact = (__bf16*)alloc((size_t)32   * 512 * 2);
  __bf16* WTcin = (__bf16*)alloc((size_t)512  * 512 * 2);
  __bf16* WTpo  = (__bf16*)alloc((size_t)1536 * 256 * 2);
  __bf16* WTpc  = (__bf16*)alloc((size_t)1536 * 512 * 2);

  const size_t BT = (size_t)B_ * T_;
  float* cin_m  = (float*)alloc(BT * 512 * 4);
  float* cout_m = (float*)alloc(BT * 512 * 4);
  float* pact   = (float*)alloc(BT * 32  * 4);
  float* obj_e  = (float*)alloc(BT * 512 * 4);
  float* cobs_e = (float*)alloc(BT * 512 * 4);
  float* act_e  = (float*)alloc(BT * 512 * 4);
  float* cout_e = (float*)alloc(BT * 512 * 4);
  float* pre0   = (float*)alloc((size_t)B_ * 512 * 4);
  float* pre1   = (float*)alloc((size_t)B_ * 512 * 4);
  float* z0     = (float*)alloc((size_t)2 * B_ * S_ * 4);   // [z0p ; z0q]
  float* z1     = (float*)alloc((size_t)2 * B_ * S_ * 4);   // [z1p ; z1q]
  float* bsum0  = (float*)alloc(512 * 4);
  float* bsum1  = (float*)alloc(512 * 4);

  auto run = [&](GemmArgs& g, int M) {
    dim3 gr((unsigned)((g.N + 255) / 256), (unsigned)(M / 16));
    gemm_wmma<<<gr, 128, 0, stream>>>(g);
  };
  const int BIG = 1 << 28;

  // ---- prep ----
  conv(Wm[0],  WTzp0,              1536, 256, 1536, 256);
  conv(Wm[1],  WTzp0 + 256 * 1536, 1536, 256, 1536, 256);
  conv(Wm[2],  WTzq0,              2560, 256, 2560, 256);
  conv(Wm[3],  WTzq0 + 256 * 2560, 2560, 256, 2560, 256);
  conv(Wm[6],  WTzp1,              512,  256, 512,  256);
  conv(Wm[7],  WTzp1 + 256 * 512,  512,  256, 512,  256);
  conv(Wm[8],  WTzq1,              1024, 256, 1024, 256);
  conv(Wm[9],  WTzq1 + 256 * 1024, 1024, 256, 1024, 256);
  conv(Wm[4],  WTx0,  768,  512, 768,  512);
  conv(Wm[5],  WTh0,  512,  512, 512,  512);
  conv(Wm[10], WTx1,  256,  512, 256,  512);
  conv(Wm[11], WTh1,  512,  512, 512,  512);
  conv(Wm[12], WTobj, 160,  512, 160,  512);
  conv(Wm[13], WTcob, 512,  512, 512,  512);
  conv(Wm[14], WTact, 16,   512, 32,   512);
  conv(Wm[15], WTcin, 512,  512, 512,  512);
  conv(Wm[16], WTpo,  1536, 160, 1536, 256);
  conv(Wm[17], WTpc,  1536, 512, 1536, 512);
  add_bias<<<2, 256, 0, stream>>>(Bv[4],  Bv[5],  bsum0, 512);   // bx0+bh0
  add_bias<<<2, 256, 0, stream>>>(Bv[10], Bv[11], bsum1, 512);   // bx1+bh1
  pad_act<<<(B_ * T_ * 32) / 256, 256, 0, stream>>>(prev_actions, pact);
  mask_comm<<<(B_ * T_ * 32) / 256, 256, 0, stream>>>(comms_in,  cin_m);
  mask_comm<<<(B_ * T_ * 32) / 256, 256, 0, stream>>>(prev_cout, cout_m);

  // ---- embeddings for all timesteps (parallel, 8192 rows) ----
  { GemmArgs g{}; g.seg[0] = Seg{objects, 0, 160, BIG, 160, WTobj, 160, 0};
    g.nseg=1; g.N=512; g.bias=Bv[12]; g.out=obj_e;  g.outS=512; g.act=1; run(g, B_*T_); }
  { GemmArgs g{}; g.seg[0] = Seg{cin_m, 0, 512, BIG, 512, WTcob, 512, 0};
    g.nseg=1; g.N=512; g.bias=Bv[13]; g.out=cobs_e; g.outS=512; g.act=1; run(g, B_*T_); }
  { GemmArgs g{}; g.seg[0] = Seg{pact, 0, 32, BIG, 32, WTact, 32, 0};
    g.nseg=1; g.N=512; g.bias=Bv[14]; g.out=act_e;  g.outS=512; g.act=1; run(g, B_*T_); }
  { GemmArgs g{}; g.seg[0] = Seg{cout_m, 0, 512, BIG, 512, WTcin, 512, 0};
    g.nseg=1; g.N=512; g.bias=Bv[15]; g.out=cout_e; g.outS=512; g.act=1; run(g, B_*T_); }

  // ---- sequential scan over T ----
  const long muRow = (long)T_ * L_ * S_;   // 32768
  const long hRow  = (long)T_ * L_ * H_;   // 65536
  const long aeRow = (long)T_ * 512;       // 32768
  for (int t = 0; t < T_; ++t) {
    const float* h0 = (t == 0) ? prevh : (hq + (size_t)(t - 1) * L_ * H_);
    const float* h1 = h0 + H_;
    const long  hS  = (t == 0) ? (long)(L_ * H_) : hRow;
    const float* ae  = act_e  + (size_t)t * 512;
    const float* ce  = cout_e + (size_t)t * 512;
    const float* oe  = obj_e  + (size_t)t * 512;
    const float* cbe = cobs_e + (size_t)t * 512;
    float* mu0p = zp_mu  + (size_t)t * L_ * S_;  float* sd0p = zp_std + (size_t)t * L_ * S_;
    float* mu0q = zq_mu  + (size_t)t * L_ * S_;  float* sd0q = zq_std + (size_t)t * L_ * S_;
    float* hpo  = hp + (size_t)t * L_ * H_;      float* hqo  = hq + (size_t)t * L_ * H_;

    // layer-0 prior heads (mu|std merged, split epilogue)
    { GemmArgs g{}; g.seg[0]=Seg{h0,0,hS,BIG,512,WTzp0,1536,0};
      g.seg[1]=Seg{ae,0,aeRow,BIG,512,WTzp0,1536,512};
      g.seg[2]=Seg{ce,0,aeRow,BIG,512,WTzp0,1536,1024};
      g.nseg=3; g.N=512; g.split=S_; g.bias=Bv[0]; g.bias2=Bv[1];
      g.out=mu0p; g.out2=sd0p; g.outS=muRow; g.act=2; g.act2=3; run(g, B_); }
    // layer-0 posterior heads
    { GemmArgs g{}; g.seg[0]=Seg{h0,0,hS,BIG,512,WTzq0,2560,0};
      g.seg[1]=Seg{oe,0,aeRow,BIG,512,WTzq0,2560,512};
      g.seg[2]=Seg{cbe,0,aeRow,BIG,512,WTzq0,2560,1024};
      g.seg[3]=Seg{ae,0,aeRow,BIG,512,WTzq0,2560,1536};
      g.seg[4]=Seg{ce,0,aeRow,BIG,512,WTzq0,2560,2048};
      g.nseg=5; g.N=512; g.split=S_; g.bias=Bv[2]; g.bias2=Bv[3];
      g.out=mu0q; g.out2=sd0q; g.outS=muRow; g.act=2; g.act2=3; run(g, B_); }
    // layer-1 prior heads
    { GemmArgs g{}; g.seg[0]=Seg{h1,0,hS,BIG,512,WTzp1,512,0};
      g.nseg=1; g.N=512; g.split=S_; g.bias=Bv[6]; g.bias2=Bv[7];
      g.out=mu0p+S_; g.out2=sd0p+S_; g.outS=muRow; g.act=2; g.act2=3; run(g, B_); }
    // pre0 = h1@Wx0[256:768] + h0@Wh0 + bx0+bh0 ; pre1 = h1@Wh1 + bx1+bh1
    { GemmArgs g{}; g.seg[0]=Seg{h1,0,hS,BIG,512,WTx0,768,256};
      g.seg[1]=Seg{h0,0,hS,BIG,512,WTh0,512,0};
      g.nseg=2; g.N=H_; g.bias=bsum0; g.out=pre0; g.outS=512; run(g, B_); }
    { GemmArgs g{}; g.seg[0]=Seg{h1,0,hS,BIG,512,WTh1,512,0};
      g.nseg=1; g.N=H_; g.bias=bsum1; g.out=pre1; g.outS=512; run(g, B_); }

    // layer-0 latents, then paired MTRNN (rows 0..127 -> hp0, 128..255 -> hq0)
    make_z2<<<(2*B_*S_)/256, 256, 0, stream>>>(mu0p, sd0p, mu0q, sd0q, muRow,
        eps_p + (size_t)(2*t) * B_ * S_, eps_q + (size_t)(2*t) * B_ * S_, z0);
    { GemmArgs g{}; g.seg[0]=Seg{z0,0,S_,BIG,S_,WTx0,768,0};
      g.nseg=1; g.N=H_; g.pairShift=7; g.outB=hpq;
      g.addmat=pre0; g.addS=512; g.blend=h0; g.blendS=hS;
      g.alpha=0.0f; g.beta=1.0f;          // tau0 = 1.0
      g.out=hpo; g.outS=hRow; g.act=2; run(g, 2*B_); }

    // layer-1 posterior heads (need hq0 of this step)
    { GemmArgs g{}; g.seg[0]=Seg{h1,0,hS,BIG,512,WTzq1,1024,0};
      g.seg[1]=Seg{hqo,0,hRow,BIG,512,WTzq1,1024,512};
      g.nseg=2; g.N=512; g.split=S_; g.bias=Bv[8]; g.bias2=Bv[9];
      g.out=mu0q+S_; g.out2=sd0q+S_; g.outS=muRow; g.act=2; g.act2=3; run(g, B_); }

    // layer-1 latents, paired MTRNN (tau1 = 4.0 -> 0.75h + 0.25 tanh)
    make_z2<<<(2*B_*S_)/256, 256, 0, stream>>>(mu0p+S_, sd0p+S_, mu0q+S_, sd0q+S_, muRow,
        eps_p + (size_t)(2*t+1) * B_ * S_, eps_q + (size_t)(2*t+1) * B_ * S_, z1);
    { GemmArgs g{}; g.seg[0]=Seg{z1,0,S_,BIG,S_,WTx1,256,0};
      g.nseg=1; g.N=H_; g.pairShift=7; g.outB=hpq;
      g.addmat=pre1; g.addS=512; g.blend=h1; g.blendS=hS;
      g.alpha=0.75f; g.beta=0.25f;
      g.out=hpo+H_; g.outS=hRow; g.act=2; run(g, 2*B_); }
  }

  // ---- predictions over (B, T-1): hwa = [hq[:, :-1, 0], act_e[:,1:], cout_e[:,1:]] ----
  { GemmArgs g{};
    g.seg[0]=Seg{hq,          hRow,  (long)(L_*H_), T_-1, 512, WTpo, 1536, 0};
    g.seg[1]=Seg{act_e + 512, aeRow, 512,           T_-1, 512, WTpo, 1536, 512};
    g.seg[2]=Seg{cout_e + 512,aeRow, 512,           T_-1, 512, WTpo, 1536, 1024};
    g.nseg=3; g.N=NOBJ_*OS_; g.bias=Bv[16]; g.out=pred_o; g.outS=NOBJ_*OS_;
    run(g, B_*(T_-1)); }
  { GemmArgs g{};
    g.seg[0]=Seg{hq,          hRow,  (long)(L_*H_), T_-1, 512, WTpc, 1536, 0};
    g.seg[1]=Seg{act_e + 512, aeRow, 512,           T_-1, 512, WTpc, 1536, 512};
    g.seg[2]=Seg{cout_e + 512,aeRow, 512,           T_-1, 512, WTpc, 1536, 1024};
    g.nseg=3; g.N=CL_*CS_; g.bias=Bv[17]; g.out=pred_c; g.outS=CL_*CS_;
    run(g, B_*(T_-1)); }
}
